// SwinTransformerBlock_72447508349163
// MI455X (gfx1250) — matrix operand order repaired and checked by reading, not compiled, not run on hardware
//
#include <hip/hip_runtime.h>
#include <hip/hip_bf16.h>
#include <cstdint>

typedef __attribute__((ext_vector_type(16))) _Float16 v16h;
typedef __attribute__((ext_vector_type(8)))  float    v8f;
typedef __attribute__((ext_vector_type(4)))  unsigned u32x4;
typedef __attribute__((ext_vector_type(8)))  int      i32x8;
typedef __attribute__((ext_vector_type(4)))  int      i32x4;

#define TOKENS   100352      // 32*56*56
#define CDIM     192
#define NHEADS   6
#define HDIM     32
#define NWIN     49          // 7*7
#define WINDOWS  2048        // 32 * 64
#define BPAD     64          // padded bias tile side

union FragH { uint4 q[2]; v16h v; };
union H8    { uint4 q; _Float16 h[8]; };

// ---------------------------------------------------------------------------
// f32 -> f16 weight conversion
// ---------------------------------------------------------------------------
__global__ void cvt_f16_kernel(const float* __restrict__ src,
                               _Float16* __restrict__ dst, int n) {
  int i = blockIdx.x * 256 + threadIdx.x;
  if (i < n) dst[i] = (_Float16)src[i];
}

// ---------------------------------------------------------------------------
// Combined rel-pos-bias + shift-mask, PADDED to 64x64 with -1e9 outside the
// valid 49x49 region so the attention kernel needs zero masking branches.
// ---------------------------------------------------------------------------
__global__ void bias_kernel(const float* __restrict__ rpb_table,
                            float* __restrict__ biasbuf) {
  int id = blockIdx.x;            // 0..383 = w2*6 + head
  int w2 = id / 6, head = id % 6;
  int wh = w2 >> 3, ww = w2 & 7;
  float* dst = biasbuf + (size_t)id * (BPAD * BPAD);
  for (int idx = threadIdx.x; idx < BPAD * BPAD; idx += blockDim.x) {
    int nq = idx >> 6, mk = idx & 63;
    float v = -1e9f;
    if (nq < NWIN && mk < NWIN) {
      int iq = nq / 7, jq = nq % 7, ik = mk / 7, jk = mk % 7;
      int rel = (iq - ik + 6) * 13 + (jq - jk + 6);
      v = rpb_table[rel * NHEADS + head];
      int hq = wh * 7 + iq, wq = ww * 7 + jq;
      int hk = wh * 7 + ik, wk = ww * 7 + jk;
      int cq = (hq < 49 ? 0 : (hq < 53 ? 1 : 2)) * 3 + (wq < 49 ? 0 : (wq < 53 ? 1 : 2));
      int ck = (hk < 49 ? 0 : (hk < 53 ? 1 : 2)) * 3 + (wk < 49 ? 0 : (wk < 53 ? 1 : 2));
      if (cq != ck) v += -100.0f;
    }
    dst[idx] = v;
  }
}

// ---------------------------------------------------------------------------
// LN1 fused with cyclic shift + window partition.  One wave per output row.
// ---------------------------------------------------------------------------
__global__ __launch_bounds__(128) void ln1_win_kernel(
    const float* __restrict__ x, const float* __restrict__ g,
    const float* __restrict__ bta, _Float16* __restrict__ xw) {
  int row  = blockIdx.x * 4 + (threadIdx.x >> 5);
  int lane = threadIdx.x & 31;
  int win = row / NWIN, n = row - win * NWIN;
  int b = win >> 6, w2 = win & 63;
  int wh = w2 >> 3, ww = w2 & 7;
  int h = (wh * 7 + n / 7 + 3) % 56;
  int w = (ww * 7 + n % 7 + 3) % 56;
  const float* src = x + ((size_t)(b * 56 + h) * 56 + w) * CDIM;
  float v[6]; float s = 0.f;
#pragma unroll
  for (int i = 0; i < 6; i++) { v[i] = src[lane + 32 * i]; s += v[i]; }
#pragma unroll
  for (int off = 16; off > 0; off >>= 1) s += __shfl_xor(s, off, 32);
  float mu = s * (1.0f / CDIM);
  float vs = 0.f;
#pragma unroll
  for (int i = 0; i < 6; i++) { float d = v[i] - mu; vs += d * d; }
#pragma unroll
  for (int off = 16; off > 0; off >>= 1) vs += __shfl_xor(vs, off, 32);
  float rstd = rsqrtf(vs * (1.0f / CDIM) + 1e-5f);
  _Float16* dst = xw + (size_t)row * CDIM;
#pragma unroll
  for (int i = 0; i < 6; i++) {
    int c = lane + 32 * i;
    dst[c] = (_Float16)((v[i] - mu) * rstd * g[c] + bta[c]);
  }
}

// ---------------------------------------------------------------------------
// LN2 in natural token order: x1 f32 -> xn2 f16
// ---------------------------------------------------------------------------
__global__ __launch_bounds__(128) void ln2_kernel(
    const float* __restrict__ x1, const float* __restrict__ g,
    const float* __restrict__ bta, _Float16* __restrict__ xn2) {
  int row  = blockIdx.x * 4 + (threadIdx.x >> 5);
  int lane = threadIdx.x & 31;
  const float* src = x1 + (size_t)row * CDIM;
  float v[6]; float s = 0.f;
#pragma unroll
  for (int i = 0; i < 6; i++) { v[i] = src[lane + 32 * i]; s += v[i]; }
#pragma unroll
  for (int off = 16; off > 0; off >>= 1) s += __shfl_xor(s, off, 32);
  float mu = s * (1.0f / CDIM);
  float vs = 0.f;
#pragma unroll
  for (int i = 0; i < 6; i++) { float d = v[i] - mu; vs += d * d; }
#pragma unroll
  for (int off = 16; off > 0; off >>= 1) vs += __shfl_xor(vs, off, 32);
  float rstd = rsqrtf(vs * (1.0f / CDIM) + 1e-5f);
  _Float16* dst = xn2 + (size_t)row * CDIM;
#pragma unroll
  for (int i = 0; i < 6; i++) {
    int c = lane + 32 * i;
    dst[c] = (_Float16)((v[i] - mu) * rstd * g[c] + bta[c]);
  }
}

// ---------------------------------------------------------------------------
// Tiled WMMA GEMM, block tile 128x64, K-step 32.  8 waves as 4x2, each wave
// 32x32 (2x2 fragments).  The A tile (128x32 f16) is staged by the Tensor
// Data Mover: one tensor_load_to_lds per K-step, with D# LDS padding
// (16 dwords data + 4 dwords pad) reproducing the 40-half LDS row stride.
// B tile staged with vector loads + global_prefetch_b8 for the next K-step.
// MODE 0: f16 out (QKV)    MODE 1: GELU+f16 (fc1)
// MODE 2: +resid f32 (fc2) MODE 3: proj scatter (window reverse+roll+shortcut)
// ---------------------------------------------------------------------------
template <int MODE>
__global__ __launch_bounds__(256, 2) void gemm_kernel(
    const _Float16* __restrict__ A, const _Float16* __restrict__ W,
    const _Float16* __restrict__ bias, int K, int N,
    const float* __restrict__ resid, float* __restrict__ outf,
    _Float16* __restrict__ outh) {
  __shared__ __align__(32) _Float16 Alds[128][40];   // 80B row stride (16B mult)
  __shared__ __align__(32) _Float16 Blds[32][80];    // 160B row stride (32B mult)
  int tid = threadIdx.x;
  int lane = tid & 31, wid = tid >> 5;
  int wm = wid >> 1, wn = wid & 1;
  int m0 = blockIdx.y * 128;
  int n0 = blockIdx.x * 64;
  unsigned ldsA = (unsigned)(unsigned long long)
      (__attribute__((address_space(3))) _Float16*)&Alds[0][0];
  v8f acc[2][2] = {};
  for (int k0 = 0; k0 < K; k0 += 32) {
    // --- A tile via Tensor Data Mover (wave 0 issues, DMA fills LDS) ---
    if (wid == 0) {
      unsigned long long ga =
          (unsigned long long)(uintptr_t)(A + (size_t)m0 * K + k0);
      u32x4 g0;
      g0[0] = 1u;                                           // count=1
      g0[1] = ldsA;                                         // lds_addr
      g0[2] = (unsigned)ga;                                 // global_addr lo
      g0[3] = (unsigned)((ga >> 32) & 0x01FFFFFFu) | (2u << 30);  // hi | type=2
      i32x8 g1;
      g1[0] = (1 << 16)      // data_size = 2 bytes
            | (1 << 20)      // pad_enable
            | (3 << 22)      // pad_interval code 3 = 16 dwords
            | (3 << 25);     // pad_amount  code 3 = 4 dwords
      g1[1] = (int)(((unsigned)K & 0xFFFFu) << 16);         // tensor_dim0 lo16
      g1[2] = (int)(((unsigned)K >> 16) |
                    (((unsigned)TOKENS & 0xFFFFu) << 16));  // dim0 hi | dim1 lo
      g1[3] = (int)(((unsigned)TOKENS >> 16) | (32u << 16)); // dim1 hi | tile_dim0
      g1[4] = 128;                                          // tile_dim1 (tile_dim2=0)
      g1[5] = K;                                            // tensor_dim0_stride lo
      g1[6] = 0;
      g1[7] = 0;
      i32x4 gz = {0, 0, 0, 0};
#if defined(__clang_major__) && (__clang_major__ >= 23)
      i32x8 gz8 = {0, 0, 0, 0, 0, 0, 0, 0};
      __builtin_amdgcn_tensor_load_to_lds(g0, g1, gz, gz, gz8, 0);
#else
      __builtin_amdgcn_tensor_load_to_lds(g0, g1, gz, gz, 0);
#endif
      __builtin_amdgcn_s_wait_tensorcnt((short)0);
    }
    // --- B tile 32x64 staged by all threads (512 uint2 chunks) ---
    for (int c = tid; c < 512; c += 256) {
      int r = c >> 4, nc = (c & 15) << 2;
      *(uint2*)&Blds[r][nc] =
          *(const uint2*)(W + (size_t)(k0 + r) * N + n0 + nc);
    }
    if (k0 + 32 < K && tid < 128)
      __builtin_prefetch(W + (size_t)(k0 + 32 + (tid >> 2)) * N + n0 + (tid & 3) * 16, 0, 1);
    __syncthreads();
    FragH af[2]; v16h bf[2];
#pragma unroll
    for (int mi = 0; mi < 2; mi++) {
      int r = wm * 32 + mi * 16 + (lane & 15);
      int kb = (lane >> 4) << 3;
      af[mi].q[0] = *(const uint4*)&Alds[r][kb];
      af[mi].q[1] = *(const uint4*)&Alds[r][16 + kb];
    }
#pragma unroll
    for (int ni = 0; ni < 2; ni++)
      for (int s = 0; s < 16; s++)
        bf[ni][s] = Blds[((s >> 3) << 4) + ((lane >> 4) << 3) + (s & 7)][wn * 32 + ni * 16 + (lane & 15)];
#pragma unroll
    for (int mi = 0; mi < 2; mi++)
#pragma unroll
      for (int ni = 0; ni < 2; ni++)
        acc[mi][ni] = __builtin_amdgcn_wmma_f32_16x16x32_f16(
            false, af[mi].v, false, bf[ni], (short)0, acc[mi][ni], false, false);
    __syncthreads();
  }
  // epilogue
#pragma unroll
  for (int mi = 0; mi < 2; mi++)
#pragma unroll
    for (int ni = 0; ni < 2; ni++) {
      int gn = n0 + wn * 32 + ni * 16 + (lane & 15);
      float bv = (float)bias[gn];
#pragma unroll
      for (int v = 0; v < 8; v++) {
        int gm = m0 + wm * 32 + mi * 16 + ((lane >> 4) << 3) + v;
        float val = acc[mi][ni][v] + bv;
        if (MODE == 0) {
          outh[(size_t)gm * N + gn] = (_Float16)val;
        } else if (MODE == 1) {
          float ge = 0.5f * val * (1.0f + erff(val * 0.70710678118f));
          outh[(size_t)gm * N + gn] = (_Float16)ge;
        } else if (MODE == 2) {
          size_t idx = (size_t)gm * N + gn;
          outf[idx] = val + resid[idx];
        } else {  // MODE 3: proj -> window reverse + roll(+3) + shortcut add
          int win = gm / NWIN, n = gm - win * NWIN;
          int b = win >> 6, w2 = win & 63;
          int wh = w2 >> 3, ww = w2 & 7;
          int h = (wh * 7 + n / 7 + 3) % 56;
          int w = (ww * 7 + n % 7 + 3) % 56;
          size_t dst = ((size_t)(b * 56 + h) * 56 + w) * CDIM + gn;
          outf[dst] = val + resid[dst];
        }
      }
    }
}

// ---------------------------------------------------------------------------
// Windowed attention, one wave per (window, head), 49 padded to 64.
// scores = (Q*scale) K^T + paddedBias   (bias is the WMMA C-matrix, from L2)
// softmax branch-free over 64 cols; out = P V.  All matmuls WMMA.
// ---------------------------------------------------------------------------
__global__ __launch_bounds__(32) void attn_kernel(
    const _Float16* __restrict__ qkv, const float* __restrict__ biasbuf,
    _Float16* __restrict__ aout) {
  __shared__ __align__(32) _Float16 Qs[64][40];   // A-frags: 16B chunks
  __shared__ __align__(32) _Float16 Kts[32][80];  // K transposed: [k][n], 32B rows
  __shared__ __align__(32) _Float16 Vs[64][48];   // [k][d], 32B rows
  __shared__ __align__(32) float    Ss[64][68];
  __shared__ __align__(32) _Float16 Ps[64][72];   // 144B rows (16B mult)
  int lane = threadIdx.x;
  int win = blockIdx.x / NHEADS, head = blockIdx.x % NHEADS;
  const float scale = 0.17677669529663687f;  // 32^-0.5
  const _Float16* base = qkv + (size_t)win * NWIN * 576 + head * HDIM;
  for (int r = lane; r < 64; r += 32) {
    if (r < NWIN) {
      const _Float16* p = base + (size_t)r * 576;
#pragma unroll
      for (int c8 = 0; c8 < 4; c8++) {
        H8 q8, k8, v8x;
        q8.q  = *(const uint4*)(p + c8 * 8);
        k8.q  = *(const uint4*)(p + 192 + c8 * 8);
        v8x.q = *(const uint4*)(p + 384 + c8 * 8);
        *(uint4*)&Vs[r][c8 * 8] = v8x.q;
#pragma unroll
        for (int i = 0; i < 8; i++) {
          Qs[r][c8 * 8 + i] = (_Float16)((float)q8.h[i] * scale);
          Kts[c8 * 8 + i][r] = k8.h[i];
        }
      }
    } else {
#pragma unroll
      for (int c = 0; c < 32; c++) {
        Qs[r][c] = (_Float16)0; Kts[c][r] = (_Float16)0; Vs[r][c] = (_Float16)0;
      }
    }
  }
  __syncthreads();
  const float* bwin =
      biasbuf + (size_t)((win & 63) * NHEADS + head) * (BPAD * BPAD);
  // scores: 4x4 tiles of 16x16, K=32, C = padded bias (L2-resident)
  for (int mt = 0; mt < 4; mt++) {
    FragH a;
    int r = mt * 16 + (lane & 15);
    int kb = (lane >> 4) << 3;
    a.q[0] = *(const uint4*)&Qs[r][kb];
    a.q[1] = *(const uint4*)&Qs[r][16 + kb];
    for (int nt = 0; nt < 4; nt++) {
      v16h b;
      for (int s = 0; s < 16; s++)
        b[s] = Kts[((s >> 3) << 4) + ((lane >> 4) << 3) + (s & 7)][nt * 16 + (lane & 15)];
      int gn = nt * 16 + (lane & 15);
      int gmb = mt * 16 + ((lane >> 4) << 3);
      v8f c;
#pragma unroll
      for (int v = 0; v < 8; v++) c[v] = bwin[(gmb + v) * BPAD + gn];
      c = __builtin_amdgcn_wmma_f32_16x16x32_f16(false, a.v, false, b, (short)0,
                                                 c, false, false);
#pragma unroll
      for (int v = 0; v < 8; v++) Ss[gmb + v][gn] = c[v];
    }
  }
  __syncthreads();
  // branch-free softmax over 64 cols (padding -> exp 0); rows < 49 only
  for (int r = lane; r < 64; r += 32) {
    if (r < NWIN) {
      float mx = -1e30f;
      for (int c = 0; c < 64; c++) mx = fmaxf(mx, Ss[r][c]);
      float sum = 0.f;
      for (int c = 0; c < 64; c++) { float e = __expf(Ss[r][c] - mx); sum += e; Ss[r][c] = e; }
      float inv = 1.0f / sum;
      for (int c = 0; c < 64; c++) Ps[r][c] = (_Float16)(Ss[r][c] * inv);
    } else {
      for (int c = 0; c < 64; c++) Ps[r][c] = (_Float16)0;
    }
  }
  __syncthreads();
  // out = P(64x64) @ V(64x32): 4x2 tiles, K in two 32-steps
  for (int mt = 0; mt < 4; mt++) {
    int rr = mt * 16 + (lane & 15);
    int kb = (lane >> 4) << 3;
    for (int nt = 0; nt < 2; nt++) {
      v8f c = {};
#pragma unroll
      for (int kk = 0; kk < 64; kk += 32) {
        FragH a;
        a.q[0] = *(const uint4*)&Ps[rr][kk + kb];
        a.q[1] = *(const uint4*)&Ps[rr][kk + 16 + kb];
        v16h b;
        for (int s = 0; s < 16; s++)
          b[s] = Vs[kk + ((s >> 3) << 4) + ((lane >> 4) << 3) + (s & 7)][nt * 16 + (lane & 15)];
        c = __builtin_amdgcn_wmma_f32_16x16x32_f16(false, a.v, false, b,
                                                   (short)0, c, false, false);
      }
      int gn = nt * 16 + (lane & 15);
#pragma unroll
      for (int v = 0; v < 8; v++) {
        int gm = mt * 16 + ((lane >> 4) << 3) + v;
        if (gm < NWIN)
          aout[(size_t)(win * NWIN + gm) * CDIM + head * HDIM + gn] = (_Float16)c[v];
      }
    }
  }
}

// ---------------------------------------------------------------------------
extern "C" void kernel_launch(void* const* d_in, const int* in_sizes, int n_in,
                              void* d_out, int out_size, void* d_ws, size_t ws_size,
                              hipStream_t stream) {
  const float* x        = (const float*)d_in[0];
  const float* norm1_g  = (const float*)d_in[1];
  const float* norm1_b  = (const float*)d_in[2];
  const float* qkv_w    = (const float*)d_in[3];
  const float* qkv_b    = (const float*)d_in[4];
  const float* rpb      = (const float*)d_in[5];
  const float* proj_w   = (const float*)d_in[6];
  const float* proj_b   = (const float*)d_in[7];
  const float* norm2_g  = (const float*)d_in[8];
  const float* norm2_b  = (const float*)d_in[9];
  const float* fc1_w    = (const float*)d_in[10];
  const float* fc1_b    = (const float*)d_in[11];
  const float* fc2_w    = (const float*)d_in[12];
  const float* fc2_b    = (const float*)d_in[13];
  float* out = (float*)d_out;

  char* base = (char*)d_ws;
  size_t off = 0;
  auto alloc = [&](size_t bytes) -> char* {
    char* p = base + off;
    off = (off + bytes + 255) & ~(size_t)255;
    return p;
  };
  _Float16* qkvw16 = (_Float16*)alloc(110592 * 2);
  _Float16* qkvb16 = (_Float16*)alloc(576 * 2);
  _Float16* projw16 = (_Float16*)alloc(36864 * 2);
  _Float16* projb16 = (_Float16*)alloc(192 * 2);
  _Float16* fc1w16 = (_Float16*)alloc(147456 * 2);
  _Float16* fc1b16 = (_Float16*)alloc(768 * 2);
  _Float16* fc2w16 = (_Float16*)alloc(147456 * 2);
  _Float16* fc2b16 = (_Float16*)alloc(192 * 2);
  float*    biasbuf = (float*)alloc((size_t)384 * BPAD * BPAD * 4);
  _Float16* xw      = (_Float16*)alloc((size_t)TOKENS * CDIM * 2);
  _Float16* qkvbuf  = (_Float16*)alloc((size_t)TOKENS * 576 * 2);
  _Float16* attnout = (_Float16*)alloc((size_t)TOKENS * CDIM * 2);
  float*    x1      = (float*)alloc((size_t)TOKENS * CDIM * 4);
  _Float16* xn2     = (_Float16*)alloc((size_t)TOKENS * CDIM * 2);
  _Float16* hbuf    = (_Float16*)alloc((size_t)TOKENS * 768 * 2);

  auto cvt = [&](const float* s, _Float16* d, int n) {
    cvt_f16_kernel<<<(n + 255) / 256, 256, 0, stream>>>(s, d, n);
  };
  cvt(qkv_w, qkvw16, 110592);  cvt(qkv_b, qkvb16, 576);
  cvt(proj_w, projw16, 36864); cvt(proj_b, projb16, 192);
  cvt(fc1_w, fc1w16, 147456);  cvt(fc1_b, fc1b16, 768);
  cvt(fc2_w, fc2w16, 147456);  cvt(fc2_b, fc2b16, 192);

  bias_kernel<<<384, 256, 0, stream>>>(rpb, biasbuf);

  // LN1 + shift + window partition
  ln1_win_kernel<<<TOKENS / 4, 128, 0, stream>>>(x, norm1_g, norm1_b, xw);

  // QKV GEMM: [100352,192] x [192,576]
  gemm_kernel<0><<<dim3(576 / 64, TOKENS / 128), 256, 0, stream>>>(
      xw, qkvw16, qkvb16, 192, 576, nullptr, nullptr, qkvbuf);

  // windowed attention
  attn_kernel<<<WINDOWS * NHEADS, 32, 0, stream>>>(qkvbuf, biasbuf, attnout);

  // proj GEMM + window reverse + roll + shortcut residual -> x1
  gemm_kernel<3><<<dim3(192 / 64, TOKENS / 128), 256, 0, stream>>>(
      attnout, projw16, projb16, 192, 192, x, x1, nullptr);

  // LN2
  ln2_kernel<<<TOKENS / 4, 128, 0, stream>>>(x1, norm2_g, norm2_b, xn2);

  // fc1 + GELU
  gemm_kernel<1><<<dim3(768 / 64, TOKENS / 128), 256, 0, stream>>>(
      xn2, fc1w16, fc1b16, 192, 768, nullptr, nullptr, hbuf);

  // fc2 + residual -> d_out
  gemm_kernel<2><<<dim3(192 / 64, TOKENS / 128), 256, 0, stream>>>(
      hbuf, fc2w16, fc2b16, 768, 192, x1, out, nullptr);
}
